// RecurrentAttention_49778670961062
// MI455X (gfx1250) — compile-verified, matrix operand order checked by reading, not run on hardware
//
#include <hip/hip_runtime.h>

// ---------------------------------------------------------------------------
// Problem constants (match the reference)
// ---------------------------------------------------------------------------
#define D_MODEL 1024
#define N_HEAD  16
#define D_HEAD  64
#define BATCH   4
#define SEQ     1024
#define BL      (BATCH * SEQ)   // 4096 rows of activations
#define DKV     (2 * D_HEAD)    // 128 (k|v concat)
#define D2      (2 * D_MODEL)   // 2048 (head-concat width)

typedef __attribute__((ext_vector_type(16))) __bf16 v16bf;
typedef __attribute__((ext_vector_type(8)))  float  v8f;
typedef unsigned __attribute__((ext_vector_type(4))) v4u;
typedef unsigned __attribute__((ext_vector_type(8))) v8u;

// ---------------------------------------------------------------------------
// Helpers
// ---------------------------------------------------------------------------
__device__ __forceinline__ __bf16 f32_to_bf16(float f) {
  unsigned u = __builtin_bit_cast(unsigned, f);
  u += 0x7FFFu + ((u >> 16) & 1u);          // round-to-nearest-even
  unsigned short s = (unsigned short)(u >> 16);
  return __builtin_bit_cast(__bf16, s);
}

__device__ __forceinline__ v8f wmma_bf16(v16bf a, v16bf b, v8f c) {
  return __builtin_amdgcn_wmma_f32_16x16x32_bf16(
      false, a, false, b, (short)0, c, false, false);
}

// K index of half 2*i of a 16-bit 16x32 A/B fragment (CDNA5 ISA §7.12.2)
__device__ __forceinline__ int frag_kbase(int i, int lane) {
  int hi = (lane & 16) ? 8 : 0;
  return (i < 4) ? (2 * i + hi) : (16 + 2 * (i - 4) + hi);
}

// Async copy 16B global -> LDS, per-lane addresses (ISA §10 / §15.18, ASYNCcnt)
__device__ __forceinline__ void async_load_b128(unsigned lds_addr, const void* gaddr) {
  asm volatile("global_load_async_to_lds_b128 %0, %1, off"
               :: "v"(lds_addr), "v"(gaddr)
               : "memory");
}
__device__ __forceinline__ void wait_asynccnt0() {
  asm volatile("s_wait_asynccnt 0x0" ::: "memory");
}
__device__ __forceinline__ unsigned lds_addr_of(const void* p) {
  return (unsigned)(unsigned long long)p;   // low 32 bits of LDS aperture addr
}

// Tensor Data Mover: 2D tile (tile_d0 x tile_d1 elements, 2B each) from a
// row-major tensor with stride_d0 into LDS with per-row padding.
// D# bitfields per CDNA5 ISA §8.3/§8.4; groups 2/3 zero (tile_dim3 unused).
__device__ __forceinline__ void tdm_load_2d_bf16(unsigned lds_addr, const void* gptr,
                                                 unsigned tensor_d0, unsigned tensor_d1,
                                                 unsigned tile_d0, unsigned tile_d1,
                                                 unsigned stride_d0,
                                                 unsigned pad_interval_code,
                                                 unsigned pad_amount_code) {
  unsigned long long ga = (unsigned long long)gptr;
  v4u g0;
  g0.x = 1u;                                         // count=1 (valid descriptor)
  g0.y = lds_addr;                                   // lds_addr [63:32]
  g0.z = (unsigned)ga;                               // global_addr [95:64]
  g0.w = ((unsigned)(ga >> 32) & 0x01FFFFFFu) | 0x80000000u;  // addr hi | type=2
  v8u g1;
  g1.s0 = (1u << 16)                                 // data_size = 2 bytes
        | (1u << 20)                                 // pad_enable
        | (pad_interval_code << 22)                  // DWORDs between pads
        | (pad_amount_code << 25);                   // pad DWORDs - 1
  g1.s1 = (tensor_d0 & 0xFFFFu) << 16;               // tensor_dim0 [79:48] lo
  g1.s2 = (tensor_d0 >> 16) | ((tensor_d1 & 0xFFFFu) << 16);
  g1.s3 = (tensor_d1 >> 16) | (tile_d0 << 16);       // tile_dim0 [127:112]
  g1.s4 = tile_d1;                                   // tile_dim1; tile_dim2=0
  g1.s5 = stride_d0;                                 // tensor_dim0_stride lo32
  g1.s6 = 0u;
  g1.s7 = 0u;
  v4u gz = {0u, 0u, 0u, 0u};
  asm volatile("tensor_load_to_lds %0, %1, %2, %3"
               :: "s"(g0), "s"(g1), "s"(gz), "s"(gz)
               : "memory");
}

// ---------------------------------------------------------------------------
// f32 -> bf16 cast
// ---------------------------------------------------------------------------
__global__ void __launch_bounds__(256)
cast_f32_bf16_kernel(const float* __restrict__ in, __bf16* __restrict__ out, int n) {
  int i = blockIdx.x * 1024 + threadIdx.x;
#pragma unroll
  for (int j = 0; j < 4; ++j) {
    int idx = i + j * 256;
    if (idx < n) out[idx] = f32_to_bf16(in[idx]);
  }
}

// ---------------------------------------------------------------------------
// Tiled bf16 GEMM: C[M,N] = A[M,K] * B[K,N] (row-major, f32 accum).
// 128x128 block tile, K-step 32, 8 waves (4x2), wave tile 32x64.
// Double-buffered LDS tiles filled with GLOBAL_LOAD_ASYNC_TO_LDS_B128 so the
// next tile's DMA overlaps the current tile's WMMAs (ASYNCcnt pipelining).
// ---------------------------------------------------------------------------
template <bool FINAL>
__global__ void __launch_bounds__(256)
gemm_bf16_kernel(const __bf16* __restrict__ A, const __bf16* __restrict__ Bm,
                 const float* __restrict__ bias, void* __restrict__ Cv,
                 int M, int N, int K) {
  __shared__ __bf16 sA[2][128][40];   // 32 + pad
  __shared__ __bf16 sB[2][32][136];   // 128 + pad

  const int tid  = threadIdx.x;
  const int lane = tid & 31;
  const int wave = tid >> 5;
  const int wm   = wave >> 1;      // 0..3
  const int wn   = wave & 1;       // 0..1
  const int ln   = lane & 15;
  const int rowBase = blockIdx.y * 128;
  const int colBase = blockIdx.x * 128;

  auto issue_tile = [&](int buf, int k0) {
    for (int c = tid; c < 512; c += 256) {
      {
        int r = c >> 2, col = (c & 3) * 8;
        async_load_b128(lds_addr_of(&sA[buf][r][col]),
                        A + (size_t)(rowBase + r) * K + k0 + col);
      }
      {
        int r = c >> 4, col = (c & 15) * 8;
        async_load_b128(lds_addr_of(&sB[buf][r][col]),
                        Bm + (size_t)(k0 + r) * N + colBase + col);
      }
    }
  };

  v8f acc[2][4] = {};
  issue_tile(0, 0);
  int cur = 0;

  for (int k0 = 0; k0 < K; k0 += 32) {
    wait_asynccnt0();        // this wave's DMA into sX[cur] has landed
    __syncthreads();         // every wave's DMA has landed
    if (k0 + 32 < K) issue_tile(cur ^ 1, k0 + 32);
    if (k0 + 64 < K)         // nudge the tile after next toward L2
      __builtin_prefetch(A + (size_t)(rowBase + (tid >> 2)) * K + k0 + 64 + (tid & 3) * 8, 0, 0);

    v16bf af[2];
#pragma unroll
    for (int mf = 0; mf < 2; ++mf) {
      int m = wm * 32 + mf * 16 + ln;
#pragma unroll
      for (int i = 0; i < 8; ++i) {
        int kb = frag_kbase(i, lane);
        af[mf][2 * i]     = sA[cur][m][kb];
        af[mf][2 * i + 1] = sA[cur][m][kb + 1];
      }
    }
#pragma unroll
    for (int nf = 0; nf < 4; ++nf) {
      v16bf bfr;
      int n = wn * 64 + nf * 16 + ln;
#pragma unroll
      for (int i = 0; i < 8; ++i) {
        int kb = frag_kbase(i, lane);
        bfr[2 * i]     = sB[cur][kb][n];
        bfr[2 * i + 1] = sB[cur][kb + 1][n];
      }
#pragma unroll
      for (int mf = 0; mf < 2; ++mf)
        acc[mf][nf] = wmma_bf16(af[mf], bfr, acc[mf][nf]);
    }
    __syncthreads();         // all waves done reading sX[cur] before its refill
    cur ^= 1;
  }

  const int mo = (lane & 16) ? 8 : 0;
#pragma unroll
  for (int mf = 0; mf < 2; ++mf) {
#pragma unroll
    for (int nf = 0; nf < 4; ++nf) {
#pragma unroll
      for (int r = 0; r < 8; ++r) {
        int gm = rowBase + wm * 32 + mf * 16 + mo + r;
        int gn = colBase + wn * 64 + nf * 16 + ln;
        if (gm < M && gn < N) {
          float vv = acc[mf][nf][r];
          if (FINAL) {
            ((float*)Cv)[(size_t)gm * N + gn] = vv + bias[gn];
          } else {
            ((__bf16*)Cv)[(size_t)gm * N + gn] = f32_to_bf16(vv);
          }
        }
      }
    }
  }
}

// ---------------------------------------------------------------------------
// Flash attention (bf16 WMMA, f32 softmax). One workgroup = 128 q-rows of one
// (batch, head). K/V tiles (64 keys x 64 dims, row stride 128 elem) are staged
// into padded LDS by the Tensor Data Mover: one wave issues the descriptor,
// waits on TENSORcnt, barrier releases the block.
// ---------------------------------------------------------------------------
__global__ void __launch_bounds__(256)
attn_kernel(const __bf16* __restrict__ Q,   // [BL, D_MODEL], head h at col h*64
            const __bf16* __restrict__ KV,  // [BL, 128]
            __bf16* __restrict__ Out,       // [BL, 2048]
            int outOff) {                   // 0 or 64 within the head slot
  __shared__ __bf16 sK[64][72];
  __shared__ __bf16 sV[64][72];
  __shared__ __bf16 sP[8][16][72];

  const int tid  = threadIdx.x;
  const int lane = tid & 31;
  const int wave = tid >> 5;
  const int ln   = lane & 15;
  const int mo   = (lane & 16) ? 8 : 0;

  const int b    = blockIdx.z;
  const int h    = blockIdx.y;
  const int qblk = blockIdx.x * 128;

  const __bf16* qbase = Q + (size_t)b * SEQ * D_MODEL + h * D_HEAD;
  const __bf16* kbase = KV + (size_t)b * SEQ * DKV;
  const __bf16* vbase = kbase + D_HEAD;

  // Q fragments: rows [qblk+wave*16, +16), dims 0..63
  v16bf qf[2];
  {
    const __bf16* qrow = qbase + (size_t)(qblk + wave * 16 + ln) * D_MODEL;
#pragma unroll
    for (int kf = 0; kf < 2; ++kf)
#pragma unroll
      for (int i = 0; i < 8; ++i) {
        int kb = frag_kbase(i, lane) + kf * 32;
        qf[kf][2 * i]     = qrow[kb];
        qf[kf][2 * i + 1] = qrow[kb + 1];
      }
  }

  float mrow[8], lrow[8];
#pragma unroll
  for (int r = 0; r < 8; ++r) { mrow[r] = -3.0e38f; lrow[r] = 0.f; }
  v8f oacc[4] = {};
  const float scale = 0.125f;  // 1/sqrt(D_HEAD)

  for (int kb0 = 0; kb0 < SEQ; kb0 += 64) {
    // TDM: 64x64-elem tiles, source row stride 128 elems; LDS rows padded
    // 128B data (32 DW, interval code 4) + 16B pad (4 DW, amount code 3) = 144B.
    if (wave == 0) {
      tdm_load_2d_bf16(lds_addr_of(&sK[0][0]),
                       kbase + (size_t)kb0 * DKV,
                       /*tensor_d0=*/DKV, /*tensor_d1=*/SEQ,
                       /*tile_d0=*/64, /*tile_d1=*/64,
                       /*stride_d0=*/DKV, /*pad_int=*/4u, /*pad_amt=*/3u);
      tdm_load_2d_bf16(lds_addr_of(&sV[0][0]),
                       vbase + (size_t)kb0 * DKV,
                       DKV, SEQ, 64, 64, DKV, 4u, 3u);
      __builtin_amdgcn_s_wait_tensorcnt(0);
    }
    __syncthreads();

    // S(16x64) = Q * K^T
    v8f s[4] = {};
#pragma unroll
    for (int nf = 0; nf < 4; ++nf) {
#pragma unroll
      for (int kf = 0; kf < 2; ++kf) {
        v16bf bfr;
        int n = nf * 16 + ln;                       // key index in tile
#pragma unroll
        for (int i = 0; i < 8; ++i) {
          int kb = frag_kbase(i, lane) + kf * 32;   // dim index
          bfr[2 * i]     = sK[n][kb];
          bfr[2 * i + 1] = sK[n][kb + 1];
        }
        s[nf] = wmma_bf16(qf[kf], bfr, s[nf]);
      }
    }

    // online softmax; row m = mo + r, stats shared across each 16-lane half
#pragma unroll
    for (int r = 0; r < 8; ++r) {
      float rm = fmaxf(fmaxf(s[0][r], s[1][r]), fmaxf(s[2][r], s[3][r]));
#pragma unroll
      for (int o = 1; o < 16; o <<= 1) rm = fmaxf(rm, __shfl_xor(rm, o, 32));
      rm *= scale;
      float nm   = fmaxf(mrow[r], rm);
      float corr = __expf(mrow[r] - nm);
      float psum = 0.f;
#pragma unroll
      for (int nf = 0; nf < 4; ++nf) {
        float pv = __expf(s[nf][r] * scale - nm);
        sP[wave][mo + r][nf * 16 + ln] = f32_to_bf16(pv);
        psum += pv;
      }
#pragma unroll
      for (int o = 1; o < 16; o <<= 1) psum += __shfl_xor(psum, o, 32);
      lrow[r] = lrow[r] * corr + psum;
      mrow[r] = nm;
#pragma unroll
      for (int f = 0; f < 4; ++f) oacc[f][r] = oacc[f][r] * corr;
    }

    // this wave's P stores must land before its cross-lane fragment reads
    asm volatile("s_wait_dscnt 0x0" ::: "memory");

    // P fragments (16 rows x 64 keys)
    v16bf pf[2];
#pragma unroll
    for (int kf = 0; kf < 2; ++kf)
#pragma unroll
      for (int i = 0; i < 8; ++i) {
        int kb = frag_kbase(i, lane) + kf * 32;     // key index
        pf[kf][2 * i]     = sP[wave][ln][kb];
        pf[kf][2 * i + 1] = sP[wave][ln][kb + 1];
      }

    // O(16x64) += P * V
#pragma unroll
    for (int f = 0; f < 4; ++f) {
#pragma unroll
      for (int kf = 0; kf < 2; ++kf) {
        v16bf bfr;
        int n = f * 16 + ln;                        // dim index
#pragma unroll
        for (int i = 0; i < 8; ++i) {
          int kb = frag_kbase(i, lane) + kf * 32;   // key index
          bfr[2 * i]     = sV[kb][n];
          bfr[2 * i + 1] = sV[kb + 1][n];
        }
        oacc[f] = wmma_bf16(pf[kf], bfr, oacc[f]);
      }
    }
    __syncthreads();   // all waves done with sK/sV before next TDM refill
  }

  // finalize: divide by row sums, emit bf16 into concat layout
#pragma unroll
  for (int r = 0; r < 8; ++r) {
    float inv = 1.f / lrow[r];
    size_t row = (size_t)b * SEQ + qblk + wave * 16 + mo + r;
    __bf16* o = Out + row * D2 + h * 128 + outOff;
#pragma unroll
    for (int f = 0; f < 4; ++f) o[f * 16 + ln] = f32_to_bf16(oacc[f][r] * inv);
  }
}

// ---------------------------------------------------------------------------
// Host orchestration
// ---------------------------------------------------------------------------
extern "C" void kernel_launch(void* const* d_in, const int* in_sizes, int n_in,
                              void* d_out, int out_size, void* d_ws, size_t ws_size,
                              hipStream_t stream) {
  (void)in_sizes; (void)n_in; (void)out_size; (void)ws_size;

  const float* qx      = (const float*)d_in[0];
  const float* kvx     = (const float*)d_in[1];
  const float* qs      = (const float*)d_in[2];
  const float* kvs     = (const float*)d_in[3];
  const float* w_qx1   = (const float*)d_in[4];
  const float* w_qs1   = (const float*)d_in[5];
  const float* w_qx2   = (const float*)d_in[6];
  const float* w_qs2   = (const float*)d_in[7];
  const float* w_kvx   = (const float*)d_in[8];
  const float* w_kvs   = (const float*)d_in[9];
  const float* w_xproj = (const float*)d_in[10];
  const float* b_xproj = (const float*)d_in[11];
  const float* w_sproj = (const float*)d_in[12];
  const float* b_sproj = (const float*)d_in[13];

  __bf16* p = (__bf16*)d_ws;
  auto take = [&](size_t n) { __bf16* r = p; p += n; return r; };
  __bf16* bqx  = take((size_t)BL * D_MODEL);
  __bf16* bqs  = take((size_t)BL * D_MODEL);
  __bf16* bkvx = take((size_t)BL * D_MODEL);
  __bf16* bkvs = take((size_t)BL * D_MODEL);
  __bf16* wq1x = take((size_t)D_MODEL * D_MODEL);
  __bf16* wq1s = take((size_t)D_MODEL * D_MODEL);
  __bf16* wq2x = take((size_t)D_MODEL * D_MODEL);
  __bf16* wq2s = take((size_t)D_MODEL * D_MODEL);
  __bf16* wkvx = take((size_t)D_MODEL * DKV);
  __bf16* wkvs = take((size_t)D_MODEL * DKV);
  __bf16* wxp  = take((size_t)D2 * D_MODEL);
  __bf16* wsp  = take((size_t)D2 * D_MODEL);
  __bf16* q1x  = take((size_t)BL * D_MODEL);
  __bf16* q1s  = take((size_t)BL * D_MODEL);
  __bf16* q2x  = take((size_t)BL * D_MODEL);
  __bf16* q2s  = take((size_t)BL * D_MODEL);
  __bf16* kvxo = take((size_t)BL * DKV);
  __bf16* kvso = take((size_t)BL * DKV);
  __bf16* catx = take((size_t)BL * D2);
  __bf16* cats = take((size_t)BL * D2);

  auto cast = [&](const float* src, __bf16* dst, int n) {
    cast_f32_bf16_kernel<<<(n + 1023) / 1024, 256, 0, stream>>>(src, dst, n);
  };
  cast(qx,      bqx,  BL * D_MODEL);
  cast(qs,      bqs,  BL * D_MODEL);
  cast(kvx,     bkvx, BL * D_MODEL);
  cast(kvs,     bkvs, BL * D_MODEL);
  cast(w_qx1,   wq1x, D_MODEL * D_MODEL);
  cast(w_qs1,   wq1s, D_MODEL * D_MODEL);
  cast(w_qx2,   wq2x, D_MODEL * D_MODEL);
  cast(w_qs2,   wq2s, D_MODEL * D_MODEL);
  cast(w_kvx,   wkvx, D_MODEL * DKV);
  cast(w_kvs,   wkvs, D_MODEL * DKV);
  cast(w_xproj, wxp,  D2 * D_MODEL);
  cast(w_sproj, wsp,  D2 * D_MODEL);

  auto gemm = [&](const __bf16* A, const __bf16* B, __bf16* C, int M, int N, int K) {
    dim3 g(N / 128, M / 128);
    gemm_bf16_kernel<false><<<g, 256, 0, stream>>>(A, B, nullptr, (void*)C, M, N, K);
  };
  gemm(bqx,  wq1x, q1x,  BL, D_MODEL, D_MODEL);
  gemm(bqs,  wq1s, q1s,  BL, D_MODEL, D_MODEL);
  gemm(bqx,  wq2x, q2x,  BL, D_MODEL, D_MODEL);
  gemm(bqs,  wq2s, q2s,  BL, D_MODEL, D_MODEL);
  gemm(bkvx, wkvx, kvxo, BL, DKV,     D_MODEL);
  gemm(bkvs, wkvs, kvso, BL, DKV,     D_MODEL);

  dim3 ga(SEQ / 128, N_HEAD, BATCH);
  attn_kernel<<<ga, 256, 0, stream>>>(q1x, kvxo, catx, 64);  // x
  attn_kernel<<<ga, 256, 0, stream>>>(q1s, kvso, cats, 64);  // s
  attn_kernel<<<ga, 256, 0, stream>>>(q2x, kvso, catx, 0);   // xs
  attn_kernel<<<ga, 256, 0, stream>>>(q2s, kvxo, cats, 0);   // sx

  dim3 gf(D_MODEL / 128, BL / 128);
  gemm_bf16_kernel<true><<<gf, 256, 0, stream>>>(catx, wxp, b_xproj, d_out, BL, D_MODEL, D2);
  gemm_bf16_kernel<true><<<gf, 256, 0, stream>>>(
      cats, wsp, b_sproj, (void*)((float*)d_out + (size_t)BL * D_MODEL), BL, D_MODEL, D2);
}
